// AutoQuantConv2d_58059367907507
// MI455X (gfx1250) — 8-bit matrix operand order repaired and checked by reading, compiled, not run on hardware
//
#include <hip/hip_runtime.h>


typedef __attribute__((ext_vector_type(16))) int      v16i;
typedef __attribute__((ext_vector_type(4)))  int      v4i;
typedef __attribute__((ext_vector_type(8)))  float    v8f;
typedef __attribute__((ext_vector_type(4)))  float    v4f;
typedef __attribute__((ext_vector_type(4)))  unsigned int v4u;

#define CIN   256
#define COUT  256
#define HH    64
#define WWID  64
#define KTOT  2304
#define BM    128
#define BN    64
#define BK    128
#define NITER (KTOT / BK)
#define ARP   144
#define BRP   144


#if __has_builtin(__builtin_amdgcn_global_load_async_to_lds_b128)
#define USE_ASYNC 1
#else
#define USE_ASYNC 0
#endif

#if USE_ASYNC
__device__ __forceinline__ void async_copy16(const unsigned char* g, unsigned char* l) {
  __builtin_amdgcn_global_load_async_to_lds_b128(
      (__attribute__((address_space(1))) v4i*)(unsigned long long)(const void*)g,
      (__attribute__((address_space(3))) v4i*)(unsigned long long)(void*)l,
      0, 0);
}
#if __has_builtin(__builtin_amdgcn_s_wait_asynccnt)
#define ASYNC_WAIT() __builtin_amdgcn_s_wait_asynccnt(0)
#else
#define ASYNC_WAIT() asm volatile("s_wait_asynccnt 0" ::: "memory")
#endif
#else
#define ASYNC_WAIT() ((void)0)
#endif


__device__ __forceinline__ float e2m1_mag(float a) {
  return (a < 0.25f) ? 0.0f
       : (a < 0.75f) ? 0.5f
       : (a < 1.25f) ? 1.0f
       : (a < 1.75f) ? 1.5f
       : (a < 2.5f)  ? 2.0f
       : (a < 3.5f)  ? 3.0f
       : (a < 5.0f)  ? 4.0f
       :               6.0f;
}

__device__ __forceinline__ float block_scale(float amax) {
  if (!(amax > 0.0f)) return 1.0f;
  int e;
  float m = frexpf(amax, &e);
  int ex = e - 3 + (m > 0.75f ? 1 : 0);
  return ldexpf(1.0f, ex);
}

__device__ __forceinline__ unsigned char f32_to_fp8_exact(float q) {
  unsigned int b = __float_as_uint(q);
  if ((b & 0x7fffffffu) == 0u) return (unsigned char)0;
  unsigned int s = (b >> 24) & 0x80u;
  int E = (int)((b >> 23) & 0xffu) - 127 + 7;
  unsigned int m = (b >> 20) & 7u;
  unsigned int mag;
  if (E >= 1) {
    mag = ((unsigned int)E << 3) | m;
  } else {
    int sh = 1 - E;
    unsigned int mf = 8u | m;
    mag = (sh < 4) ? (mf >> sh) : 0u;
  }
  return (unsigned char)(s | mag);
}

__device__ __forceinline__ void quant_block16_fp8(const float* __restrict__ v,
                                                  unsigned char* __restrict__ h) {
  float amax = 0.0f;
#pragma unroll
  for (int i = 0; i < 16; ++i) amax = fmaxf(amax, fabsf(v[i]));
  float scale = block_scale(amax);
  float inv   = 1.0f / scale;
#pragma unroll
  for (int i = 0; i < 16; ++i) {
    float u = v[i] * inv;
    float g = e2m1_mag(fabsf(u));
    float q = (u < 0.0f) ? -g : g;
    h[i] = f32_to_fp8_exact(q * scale);
  }
}


__global__ __launch_bounds__(256) void quant_x_fp8_kernel(
    const float* __restrict__ x, unsigned char* __restrict__ qx) {
  __shared__ unsigned char T[256 * 80];
  const int p = blockIdx.x;
  const int n = p >> 6;
  const int y = p & 63;
  const int c = threadIdx.x;

  const float* src = x + (((size_t)n * CIN + c) * HH + y) * WWID;
  float vals[64];
#pragma unroll
  for (int j = 0; j < 16; ++j) {
    v4f t;
    __builtin_memcpy(&t, src + 4 * j, 16);
#pragma unroll
    for (int k = 0; k < 4; ++k) vals[4 * j + k] = t[k];
  }
  unsigned char q8[64];
#pragma unroll
  for (int b = 0; b < 4; ++b) quant_block16_fp8(&vals[b * 16], &q8[b * 16]);

#pragma unroll
  for (int j = 0; j < 4; ++j) {
    v4u wv;
#pragma unroll
    for (int k = 0; k < 4; ++k) {
      int o = j * 16 + k * 4;
      wv[k] = (unsigned)q8[o] | ((unsigned)q8[o + 1] << 8) |
              ((unsigned)q8[o + 2] << 16) | ((unsigned)q8[o + 3] << 24);
    }
    *(v4u*)&T[c * 80 + j * 16] = wv;
  }
  __syncthreads();

  const int xp = threadIdx.x & 63;
  const int cg = (threadIdx.x >> 6) << 6;
  unsigned char* dst = qx + (((size_t)p * WWID + xp) << 8) + cg;
#pragma unroll
  for (int j = 0; j < 4; ++j) {
    v4u wv;
#pragma unroll
    for (int k = 0; k < 4; ++k) {
      int cb = cg + j * 16 + k * 4;
      wv[k] = (unsigned)T[(cb + 0) * 80 + xp] |
              ((unsigned)T[(cb + 1) * 80 + xp] << 8) |
              ((unsigned)T[(cb + 2) * 80 + xp] << 16) |
              ((unsigned)T[(cb + 3) * 80 + xp] << 24);
    }
    *(v4u*)(dst + j * 16) = wv;
  }
}


__global__ __launch_bounds__(256) void quant_w_fp8_kernel(
    const float* __restrict__ w, unsigned char* __restrict__ qwt, int nblocks) {
  int b = blockIdx.x * 256 + threadIdx.x;
  if (b >= nblocks) return;
  const float* src = w + (size_t)b * 16;
  float v[16];
#pragma unroll
  for (int i = 0; i < 16; ++i) v[i] = src[i];
  unsigned char h[16];
  quant_block16_fp8(v, h);
  int base = b * 16;
#pragma unroll
  for (int i = 0; i < 16; ++i) {
    int f   = base + i;
    int o   = f / 2304;
    int rem = f - o * 2304;
    int c   = rem / 9;
    int rs  = rem - c * 9;
    int k   = c & 127;
    int g   = k >> 3;
    int wnd = k & 7;
    int cp  = (c & 128) | (((g >> 1) & 1) << 6) | ((g >> 2) << 4) | ((g & 1) << 3) | wnd;
    qwt[(size_t)o * KTOT + rs * 256 + cp] = h[i];
  }
}


__global__ __launch_bounds__(256) void conv_wmma_fp8_kernel(
    const unsigned char* __restrict__ qx,
    const unsigned char* __restrict__ qwt,
    float* __restrict__ out)
{
  __shared__ unsigned char As[2][BM * ARP];
  __shared__ unsigned char Bs[2][BN * BRP];

  const int tid  = threadIdx.x;
  const int lane = tid & 31;
  const int wave = tid >> 5;
  const int wm   = wave >> 1;
  const int wn   = wave & 1;
  const int lh   = lane >> 4;
  const int l16  = lane & 15;

  const int tile = blockIdx.x;
  const int tm   = tile >> 11;
  const int prow = tile & 2047;
  const int n    = prow >> 6;
  const int y    = prow & 63;
  const int m0   = tm * BM;

  const int s_nn = tid >> 2;
  const int s_ch = tid & 3;

  v8f acc00 = {}, acc01 = {}, acc10 = {}, acc11 = {};

  auto stage = [&](int kk0, int buf) {
    const int rs = kk0 >> 8;
    const int c0 = kk0 & 255;
    const int r  = rs / 3;
    const int s  = rs - r * 3;
    const int yy = y + r - 1;

#pragma unroll
    for (int p = 0; p < 4; ++p) {
      int lin = p * 256 + tid;
      int row = lin >> 3;
      int kch = (lin & 7) << 4;
      const unsigned char* g = qwt + (size_t)(m0 + row) * KTOT + kk0 + kch;
      unsigned char* l = &As[buf][row * ARP + kch];
#if USE_ASYNC
      async_copy16(g, l);
#else
      v4i v;
      __builtin_memcpy(&v, g, 16);
      *(v4i*)l = v;
#endif
    }

    {
      int xg = s_nn + s - 1;
      unsigned char* l0 = &Bs[buf][s_nn * BRP + (s_ch << 5)];
      if (yy >= 0 && yy < HH && xg >= 0 && xg < WWID) {
        const unsigned char* src =
            qx + ((((size_t)(n * HH + yy)) * WWID + xg) << 8) + c0 + (s_ch << 5);
#if USE_ASYNC
        async_copy16(src, l0);
        async_copy16(src + 16, l0 + 16);
#else
        v4i v0, v1;
        __builtin_memcpy(&v0, src, 16);
        __builtin_memcpy(&v1, src + 16, 16);
        *(v4i*)l0 = v0;
        *(v4i*)(l0 + 16) = v1;
#endif
      } else {
        v4i z = {};
        *(v4i*)l0 = z;
        *(v4i*)(l0 + 16) = z;
      }
    }
  };

  stage(0, 0);

  for (int it = 0; it < NITER; ++it) {
    ASYNC_WAIT();
    __syncthreads();
    if (it + 1 < NITER) stage((it + 1) * BK, (it + 1) & 1);

    const unsigned char* Ab = &As[it & 1][0];
    const unsigned char* Bb = &Bs[it & 1][0];

    v16i a0, a1, b0, b1;
    {
      const int base0 = (wm * 32 + l16) * ARP + lh * 64;
      const int base1 = base0 + 16 * ARP;
#pragma unroll
      for (int j = 0; j < 4; ++j) {
        v4i t0 = *(const v4i*)&Ab[base0 + 16 * j];
        v4i t1 = *(const v4i*)&Ab[base1 + 16 * j];
#pragma unroll
        for (int k = 0; k < 4; ++k) { a0[4 * j + k] = t0[k]; a1[4 * j + k] = t1[k]; }
      }
    }
    {
      const int cb0 = (wn * 32 + l16) * BRP + lh * 16;
      const int cb1 = cb0 + 16 * BRP;
#pragma unroll
      for (int j = 0; j < 4; ++j) {
        v4i t0 = *(const v4i*)&Bb[cb0 + 32 * j];
        v4i t1 = *(const v4i*)&Bb[cb1 + 32 * j];
#pragma unroll
        for (int k = 0; k < 4; ++k) { b0[4 * j + k] = t0[k]; b1[4 * j + k] = t1[k]; }
      }
    }

    acc00 = __builtin_amdgcn_wmma_f32_16x16x128_fp8_fp8(a0, b0, (short)0, acc00, false, false);
    acc01 = __builtin_amdgcn_wmma_f32_16x16x128_fp8_fp8(a0, b1, (short)0, acc01, false, false);
    acc10 = __builtin_amdgcn_wmma_f32_16x16x128_fp8_fp8(a1, b0, (short)0, acc10, false, false);
    acc11 = __builtin_amdgcn_wmma_f32_16x16x128_fp8_fp8(a1, b1, (short)0, acc11, false, false);
  }

  float* obase = out + (((size_t)n * COUT) * HH + y) * WWID;
#pragma unroll
  for (int j = 0; j < 8; ++j) {
    int mr0 = m0 + wm * 32 + lh * 8 + j;
    int mr1 = mr0 + 16;
    int cx0 = wn * 32 + l16;
    int cx1 = cx0 + 16;
    obase[(size_t)mr0 * (HH * WWID) + cx0] = acc00[j];
    obase[(size_t)mr0 * (HH * WWID) + cx1] = acc01[j];
    obase[(size_t)mr1 * (HH * WWID) + cx0] = acc10[j];
    obase[(size_t)mr1 * (HH * WWID) + cx1] = acc11[j];
  }
}


extern "C" void kernel_launch(void* const* d_in, const int* in_sizes, int n_in,
                              void* d_out, int out_size, void* d_ws, size_t ws_size,
                              hipStream_t stream) {
  const float* x = (const float*)d_in[0];
  const float* w = (const float*)d_in[1];
  float* out = (float*)d_out;

  unsigned char* qx8 = (unsigned char*)d_ws;
  unsigned char* qwt = (unsigned char*)d_ws + (size_t)33554432;

  const int wblocks = 589824 / 16;

  quant_x_fp8_kernel<<<dim3(2048), dim3(256), 0, stream>>>(x, qx8);
  quant_w_fp8_kernel<<<dim3(144), dim3(256), 0, stream>>>(w, qwt, wblocks);
  conv_wmma_fp8_kernel<<<dim3(4096), dim3(256), 0, stream>>>(qx8, qwt, out);
}
